// Decoder_46866683134303
// MI455X (gfx1250) — matrix operand order repaired and checked by reading, not compiled, not run on hardware
//
#include <hip/hip_runtime.h>
#include <hip/hip_bf16.h>
#include <cstdint>

typedef __bf16 bf16;
typedef __attribute__((ext_vector_type(16))) __bf16 v16bf;
typedef __attribute__((ext_vector_type(8)))  __bf16 v8bf;
typedef __attribute__((ext_vector_type(8)))  float  v8f;

#define S_LEN 2048
#define EMB   4096
#define NH    32
#define NKV   8
#define HD    128
#define GRP   4
#define NUM_REMAIN 410
#define NEGI  (-1.0e30f)

union FragB { v16bf v; v8bf h[2]; };

// ---- CDNA5 async global->LDS copy (ASYNCcnt-tracked, no VGPR staging) ----
__device__ __forceinline__ void async_copy16(uint32_t lds_byte_addr, uint64_t gaddr) {
  // GLOBAL_LOAD_ASYNC_TO_LDS_B128: each lane copies 16B from global address vaddr to LDS byte address vdst
  asm volatile("global_load_async_to_lds_b128 %0, %1, off"
               :: "v"(lds_byte_addr), "v"(gaddr) : "memory");
}
__device__ __forceinline__ void wait_async0() {
  asm volatile("s_wait_asynccnt 0x0" ::: "memory");
}
__device__ __forceinline__ uint32_t lds_addr32(const void* p) {
  // generic LDS pointer = {shared aperture, offset[31:0]} -> low 32 bits are the LDS byte address
  return (uint32_t)(uintptr_t)p;
}

// ---------------------------------------------------------------- converts
__global__ void k_f32_to_bf16(const float* __restrict__ src, bf16* __restrict__ dst, int n) {
  int i = blockIdx.x * blockDim.x + threadIdx.x;
  if (i < n) dst[i] = (bf16)src[i];
}

// ---------------------------------------------------------------- GEMM: C[M,N] = A[M,K] * W[N,K]^T (bf16 in, fp32 out)
// block = 256x128 tile, 256 threads = 8 waves; wave tile = 64x64 = 4x4 wmma tiles.
// Double-buffered LDS filled by global_load_async_to_lds_b128.
__global__ __launch_bounds__(256) void k_gemm_bf16(
    const bf16* __restrict__ A, const bf16* __restrict__ W,
    float* __restrict__ C, int M, int N, int K)
{
  constexpr int LDA = 40; // 32 + 8 pad elements, 80B row stride (16B aligned)
  __shared__ __align__(16) bf16 As[2][256 * LDA];
  __shared__ __align__(16) bf16 Bs[2][128 * LDA];

  const int tid  = threadIdx.x;
  const int lane = tid & 31;
  const int w    = tid >> 5;
  const int wm   = w >> 1;        // 4 wave-rows of 64
  const int wn   = w & 1;         // 2 wave-cols of 64
  const int mb   = blockIdx.y * 256;
  const int nb   = blockIdx.x * 128;
  const int hl   = lane >> 4;     // lane half (K-chunk select)
  const int l16  = lane & 15;

  const v8f vzero = {};
  v8f acc[4][4];
  for (int i = 0; i < 4; i++) for (int j = 0; j < 4; j++) acc[i][j] = vzero;

  // async staging map: A: thread t -> row t, 4x16B chunks; B: thread t -> row t>>1, 2x16B chunks
  const int brow = tid >> 1, bcol = (tid & 1) * 16;
  auto issue_tile = [&](int buf, int kc) {
    const bf16* ga = A + (size_t)(mb + tid) * K + kc;
    uint32_t la = lds_addr32(&As[buf][tid * LDA]);
    #pragma unroll
    for (int c = 0; c < 4; c++)
      async_copy16(la + c * 16, (uint64_t)(uintptr_t)(ga + c * 8));
    const bf16* gb = W + (size_t)(nb + brow) * K + kc + bcol;
    uint32_t lb = lds_addr32(&Bs[buf][brow * LDA + bcol]);
    async_copy16(lb,      (uint64_t)(uintptr_t)gb);
    async_copy16(lb + 16, (uint64_t)(uintptr_t)(gb + 8));
  };

  int buf = 0;
  issue_tile(0, 0);                      // prologue fill
  for (int kc = 0; kc < K; kc += 32) {
    wait_async0();                       // my slice of buf is in LDS
    __syncthreads();                     // everyone's slice is in LDS; prev buf free
    if (kc + 32 < K) issue_tile(buf ^ 1, kc + 32);   // async fill next buffer under compute

    FragB a[4], b[4];
    #pragma unroll
    for (int i = 0; i < 4; i++) {
      int row = wm * 64 + i * 16 + l16;  // A: lane l16 owns row, K chunks {8h.., 16+8h..}
      a[i].h[0] = *(const v8bf*)&As[buf][row * LDA + 8 * hl];
      a[i].h[1] = *(const v8bf*)&As[buf][row * LDA + 16 + 8 * hl];
    }
    #pragma unroll
    for (int j = 0; j < 4; j++) {
      int col = wn * 64 + j * 16 + l16;  // B: lane l16 owns col, K chunks {8h.., 16+8h..}
      b[j].h[0] = *(const v8bf*)&Bs[buf][col * LDA + 8 * hl];
      b[j].h[1] = *(const v8bf*)&Bs[buf][col * LDA + 16 + 8 * hl];
    }
    #pragma unroll
    for (int i = 0; i < 4; i++)
      #pragma unroll
      for (int j = 0; j < 4; j++)
        acc[i][j] = __builtin_amdgcn_wmma_f32_16x16x32_bf16(
            false, a[i].v, false, b[j].v, (short)0, acc[i][j], false, false);
    buf ^= 1;
  }

  for (int i = 0; i < 4; i++) {
    int mrow0 = mb + wm * 64 + i * 16 + hl * 8;      // D: vgpr r -> row 8h+r, col = lane&15
    for (int j = 0; j < 4; j++) {
      int col = nb + wn * 64 + j * 16 + l16;
      float* cp = C + (size_t)mrow0 * N + col;
      #pragma unroll
      for (int r = 0; r < 8; r++) cp[(size_t)r * N] = acc[i][j][r];
    }
  }
}

// ---------------------------------------------------------------- RoPE + head-major reshape ([S,h*D] f32 -> [h,S,D] bf16)
__global__ void k_rope(const float* __restrict__ x, bf16* __restrict__ out, int heads) {
  int idx = blockIdx.x * blockDim.x + threadIdx.x;
  if (idx >= heads * S_LEN * HD) return;
  int d = idx & (HD - 1);
  int s = (idx >> 7) & (S_LEN - 1);
  int h = idx >> 18;                      // / (S_LEN*HD)
  float inv = __powf(10000.0f, -(float)(2 * (d & 63)) / (float)HD);
  float si, co;
  __sincosf((float)s * inv, &si, &co);
  const float* row = x + (size_t)s * (heads * HD) + h * HD;
  float xv  = row[d];
  float rot = (d < 64) ? -row[d + 64] : row[d - 64];
  out[idx] = (bf16)(xv * co + rot * si);
}

__global__ void k_reshape_v(const float* __restrict__ x, bf16* __restrict__ out) {
  int idx = blockIdx.x * blockDim.x + threadIdx.x;
  if (idx >= NKV * S_LEN * HD) return;
  int d = idx & (HD - 1);
  int s = (idx >> 7) & (S_LEN - 1);
  int h = idx >> 18;
  out[idx] = (bf16)x[(size_t)s * (NKV * HD) + h * HD + d];
}

// ---------------------------------------------------------------- hash MLP + sign bit packing
__global__ __launch_bounds__(128) void k_hash(
    const bf16* __restrict__ qr, const bf16* __restrict__ kr,
    const float* __restrict__ P1, const float* __restrict__ B1,
    const float* __restrict__ P2, const float* __restrict__ B2,
    uint32_t* __restrict__ qm, uint32_t* __restrict__ km)
{
  __shared__ float xs[HD];
  __shared__ float ys[HD];
  const int s = blockIdx.x, h = blockIdx.y, isK = blockIdx.z;
  const int e = threadIdx.x;
  const bf16* src = isK ? (kr + ((size_t)(h / GRP) * S_LEN + s) * HD)
                        : (qr + ((size_t)h * S_LEN + s) * HD);
  xs[e] = (float)src[e];
  __syncthreads();
  const float* p1 = P1 + (size_t)h * HD * HD;
  float acc = B1[h * HD + e];
  for (int d = 0; d < HD; d++) acc = fmaf(xs[d], p1[d * HD + e], acc);
  float y = acc / (1.f + __expf(-acc)) + xs[e];      // silu residual
  ys[e] = y;
  __syncthreads();
  const float* p2 = P2 + (size_t)h * HD * HD;
  float acc2 = B2[h * HD + e];
  for (int d = 0; d < HD; d++) acc2 = fmaf(ys[d], p2[d * HD + e], acc2);
  float z = acc2 + y;
  unsigned long long bal = __ballot(z >= 0.0f);      // wave32 -> low 32 bits
  uint32_t* dst = (isK ? km : qm) + ((size_t)h * S_LEN + s) * 4 + (e >> 5);
  if ((e & 31) == 0) *dst = (uint32_t)bal;
}

// ---------------------------------------------------------------- per-row top-k threshold via 129-bin histogram
__global__ __launch_bounds__(256) void k_thresh(
    const uint32_t* __restrict__ qm, const uint32_t* __restrict__ km,
    int* __restrict__ thr)
{
  __shared__ int hist[8][130];
  const int w = threadIdx.x >> 5;
  const int lane = threadIdx.x & 31;
  const int row = blockIdx.x * 8 + w;                // 0..NH*S-1
  const int h = row >> 11;
  const int s = row & (S_LEN - 1);
  for (int b = lane; b < 130; b += 32) hist[w][b] = 0;
  if (s + 1 <= NUM_REMAIN) { if (lane == 0) thr[row] = -1000; return; } // keep all
  const uint32_t* qp = qm + (size_t)row * 4;
  uint32_t q0 = qp[0], q1 = qp[1], q2 = qp[2], q3 = qp[3];
  for (int j = lane; j <= s; j += 32) {
    const uint32_t* kp = km + ((size_t)h * S_LEN + j) * 4;
    int pc = __popc(q0 ^ kp[0]) + __popc(q1 ^ kp[1]) + __popc(q2 ^ kp[2]) + __popc(q3 ^ kp[3]);
    atomicAdd(&hist[w][(128 - 2 * pc + 128) >> 1], 1);   // sim even in [-128,128]
  }
  if (lane == 0) {
    int cum = 0, t = -1000;
    for (int b = 128; b >= 0; b--) {
      cum += hist[w][b];
      if (cum >= NUM_REMAIN) { t = 2 * b - 128; break; }
    }
    thr[row] = t;
  }
}

// ---------------------------------------------------------------- flash attention with WMMA + LSH mask
__global__ __launch_bounds__(128) void k_attn(
    const bf16* __restrict__ qr, const bf16* __restrict__ kr, const bf16* __restrict__ vb,
    const uint32_t* __restrict__ qm, const uint32_t* __restrict__ km,
    const int* __restrict__ thr, bf16* __restrict__ ao)
{
  constexpr int LK = 136;   // Ks row stride (128+8)
  constexpr int LV = 40;    // Vt row stride (32+8)
  constexpr int LP = 40;
  __shared__ __align__(16) bf16 Ks[32 * LK];
  __shared__ __align__(16) bf16 Vt[128 * LV];
  __shared__ __align__(16) bf16 Ps[4][16 * LP];

  const int h   = blockIdx.y;
  const int kvh = h >> 2;
  const int qb0 = blockIdx.x * 64;
  const int w   = threadIdx.x >> 5;
  const int lane = threadIdx.x & 31;
  const int hl  = lane >> 4, l16 = lane & 15;
  const int qb  = qb0 + w * 16;                // this wave's 16 q rows

  // Q A-fragments (4 K-chunks of 32 dims), straight from global (rows contiguous)
  FragB qa[4];
  {
    const bf16* qrow = qr + ((size_t)h * S_LEN + qb + l16) * HD;
    for (int c = 0; c < 4; c++) {
      qa[c].h[0] = *(const v8bf*)(qrow + c * 32 + 8 * hl);
      qa[c].h[1] = *(const v8bf*)(qrow + c * 32 + 16 + 8 * hl);
    }
  }
  // per-lane row metadata (rows qb + 8*hl + r)
  int trh[8]; uint32_t um[8][4];
  for (int r = 0; r < 8; r++) {
    int qrow = qb + 8 * hl + r;
    trh[r] = thr[h * S_LEN + qrow];
    const uint32_t* qp = qm + ((size_t)h * S_LEN + qrow) * 4;
    um[r][0] = qp[0]; um[r][1] = qp[1]; um[r][2] = qp[2]; um[r][3] = qp[3];
  }

  const v8f vzero = {};
  v8f Oacc[8];
  for (int t = 0; t < 8; t++) Oacc[t] = vzero;
  float mrow[8], lsum[8];
  for (int r = 0; r < 8; r++) { mrow[r] = NEGI; lsum[r] = 0.f; }
  const float scale = 0.08838834764831845f;   // 1/sqrt(128)

  const int kend = qb0 + 64;
  for (int kb = 0; kb < kend; kb += 32) {
    __syncthreads();
    { // stage K block [32 keys][128 d] and V transposed [128 d][32 keys]
      int key = threadIdx.x >> 2;
      int db  = (threadIdx.x & 3) * 32;
      const bf16* kg = kr + ((size_t)kvh * S_LEN + kb + key) * HD + db;
      for (int c = 0; c < 4; c++)
        *(v8bf*)&Ks[key * LK + db + c * 8] = *(const v8bf*)(kg + c * 8);
      const bf16* vg = vb + ((size_t)kvh * S_LEN + kb + key) * HD + db;
      for (int i = 0; i < 32; i++) Vt[(db + i) * LV + key] = vg[i];
    }
    __syncthreads();

    // S = Q K^T : two 16-key tiles
    v8f pt[2];
    for (int t = 0; t < 2; t++) {
      v8f p = vzero;
      int col = t * 16 + l16;
      for (int c = 0; c < 4; c++) {
        FragB bk;
        bk.h[0] = *(const v8bf*)&Ks[col * LK + c * 32 + 8 * hl];
        bk.h[1] = *(const v8bf*)&Ks[col * LK + c * 32 + 16 + 8 * hl];
        p = __builtin_amdgcn_wmma_f32_16x16x32_bf16(false, qa[c].v, false, bk.v,
                                                    (short)0, p, false, false);
      }
      // causal + LSH mask (sim via popcount of packed signs)
      int j = kb + t * 16 + l16;
      const uint32_t* kp = km + ((size_t)h * S_LEN + j) * 4;
      uint32_t k0 = kp[0], k1 = kp[1], k2 = kp[2], k3 = kp[3];
      #pragma unroll
      for (int r = 0; r < 8; r++) {
        int qrow = qb + 8 * hl + r;
        int sim = 128 - 2 * (__popc(um[r][0] ^ k0) + __popc(um[r][1] ^ k1) +
                             __popc(um[r][2] ^ k2) + __popc(um[r][3] ^ k3));
        bool keep = (j <= qrow) && (sim >= trh[r]);
        p[r] = keep ? p[r] * scale : NEGI;
      }
      pt[t] = p;
    }

    // online softmax update over these 32 columns (rows live in 16-lane halves)
    float mnew[8];
    for (int r = 0; r < 8; r++) {
      float v = fmaxf(pt[0][r], pt[1][r]);
      for (int m = 1; m < 16; m <<= 1) v = fmaxf(v, __shfl_xor(v, m, 32));
      mnew[r] = fmaxf(mrow[r], v);
    }
    for (int t = 0; t < 2; t++)
      #pragma unroll
      for (int r = 0; r < 8; r++) {
        float e = __expf(pt[t][r] - mnew[r]);
        pt[t][r] = e;
        Ps[w][(8 * hl + r) * LP + t * 16 + l16] = (bf16)e;   // D-layout -> A-layout via LDS
      }
    for (int r = 0; r < 8; r++) {
      float sv = pt[0][r] + pt[1][r];
      for (int m = 1; m < 16; m <<= 1) sv += __shfl_xor(sv, m, 32);
      float sc = __expf(mrow[r] - mnew[r]);
      lsum[r] = lsum[r] * sc + sv;
      mrow[r] = mnew[r];
      for (int t = 0; t < 8; t++) Oacc[t][r] *= sc;
    }

    // O += P V : A = P (16x32 keys) from per-wave Ps, B = Vt rows (keys {8h.., 16+8h..})
    FragB pa;
    pa.h[0] = *(const v8bf*)&Ps[w][l16 * LP + 8 * hl];
    pa.h[1] = *(const v8bf*)&Ps[w][l16 * LP + 16 + 8 * hl];
    for (int t = 0; t < 8; t++) {
      FragB bv;
      int drow = t * 16 + l16;
      bv.h[0] = *(const v8bf*)&Vt[drow * LV + 8 * hl];
      bv.h[1] = *(const v8bf*)&Vt[drow * LV + 16 + 8 * hl];
      Oacc[t] = __builtin_amdgcn_wmma_f32_16x16x32_bf16(false, pa.v, false, bv.v,
                                                        (short)0, Oacc[t], false, false);
    }
  }

  for (int r = 0; r < 8; r++) {
    int qrow = qb + 8 * hl + r;
    float inv = (lsum[r] > 0.f) ? 1.f / lsum[r] : 0.f;
    for (int t = 0; t < 8; t++)
      ao[(size_t)qrow * (NH * HD) + h * HD + t * 16 + l16] = (bf16)(Oacc[t][r] * inv);
  }
}

// ---------------------------------------------------------------- launcher
extern "C" void kernel_launch(void* const* d_in, const int* in_sizes, int n_in,
                              void* d_out, int out_size, void* d_ws, size_t ws_size,
                              hipStream_t stream)
{
  (void)in_sizes; (void)n_in; (void)out_size; (void)ws_size;
  const float* hidden = (const float*)d_in[0];
  const float* Wq = (const float*)d_in[1];
  const float* Wk = (const float*)d_in[2];
  const float* Wv = (const float*)d_in[3];
  const float* Wo = (const float*)d_in[4];
  const float* P1 = (const float*)d_in[5];
  const float* B1 = (const float*)d_in[6];
  const float* P2 = (const float*)d_in[7];
  const float* B2 = (const float*)d_in[8];

  char* ws = (char*)d_ws;
  size_t off = 0;
  auto alloc = [&](size_t bytes) -> char* {
    char* p = ws + off;
    off += (bytes + 255) & ~(size_t)255;
    return p;
  };
  bf16* hb   = (bf16*)alloc((size_t)S_LEN * EMB * 2);
  bf16* wqb  = (bf16*)alloc((size_t)EMB * EMB * 2);
  bf16* wkb  = (bf16*)alloc((size_t)NKV * HD * EMB * 2);
  bf16* wvb  = (bf16*)alloc((size_t)NKV * HD * EMB * 2);
  bf16* wob  = (bf16*)alloc((size_t)EMB * EMB * 2);
  float* qf  = (float*)alloc((size_t)S_LEN * EMB * 4);
  float* kf  = (float*)alloc((size_t)S_LEN * NKV * HD * 4);
  float* vf  = (float*)alloc((size_t)S_LEN * NKV * HD * 4);
  bf16* qrb  = (bf16*)alloc((size_t)NH * S_LEN * HD * 2);
  bf16* krb  = (bf16*)alloc((size_t)NKV * S_LEN * HD * 2);
  bf16* vbb  = (bf16*)alloc((size_t)NKV * S_LEN * HD * 2);
  uint32_t* qmask = (uint32_t*)alloc((size_t)NH * S_LEN * 16);
  uint32_t* kmask = (uint32_t*)alloc((size_t)NH * S_LEN * 16);
  int* thr   = (int*)alloc((size_t)NH * S_LEN * 4);
  bf16* ao   = (bf16*)alloc((size_t)S_LEN * EMB * 2);

  int n;
  n = S_LEN * EMB;    k_f32_to_bf16<<<(n + 255) / 256, 256, 0, stream>>>(hidden, hb, n);
  n = EMB * EMB;      k_f32_to_bf16<<<(n + 255) / 256, 256, 0, stream>>>(Wq, wqb, n);
  n = NKV * HD * EMB; k_f32_to_bf16<<<(n + 255) / 256, 256, 0, stream>>>(Wk, wkb, n);
  n = NKV * HD * EMB; k_f32_to_bf16<<<(n + 255) / 256, 256, 0, stream>>>(Wv, wvb, n);
  n = EMB * EMB;      k_f32_to_bf16<<<(n + 255) / 256, 256, 0, stream>>>(Wo, wob, n);

  k_gemm_bf16<<<dim3(EMB / 128, S_LEN / 256), 256, 0, stream>>>(hb, wqb, qf, S_LEN, EMB, EMB);
  k_gemm_bf16<<<dim3(NKV * HD / 128, S_LEN / 256), 256, 0, stream>>>(hb, wkb, kf, S_LEN, NKV * HD, EMB);
  k_gemm_bf16<<<dim3(NKV * HD / 128, S_LEN / 256), 256, 0, stream>>>(hb, wvb, vf, S_LEN, NKV * HD, EMB);

  n = NH * S_LEN * HD;  k_rope<<<(n + 255) / 256, 256, 0, stream>>>(qf, qrb, NH);
  n = NKV * S_LEN * HD; k_rope<<<(n + 255) / 256, 256, 0, stream>>>(kf, krb, NKV);
  n = NKV * S_LEN * HD; k_reshape_v<<<(n + 255) / 256, 256, 0, stream>>>(vf, vbb);

  k_hash<<<dim3(S_LEN, NH, 2), 128, 0, stream>>>(qrb, krb, P1, B1, P2, B2, qmask, kmask);
  k_thresh<<<dim3(NH * S_LEN / 8), 256, 0, stream>>>(qmask, kmask, thr);
  k_attn<<<dim3(S_LEN / 64, NH), 128, 0, stream>>>(qrb, krb, vbb, qmask, kmask, thr, ao);

  k_gemm_bf16<<<dim3(EMB / 128, S_LEN / 256), 256, 0, stream>>>(ao, wob, (float*)d_out, S_LEN, EMB, EMB);
}
